// PQMF_52974126628914
// MI455X (gfx1250) — matrix operand order repaired and checked by reading, not compiled, not run on hardware
//
#include <hip/hip_runtime.h>
#include <hip/hip_bf16.h>

typedef __bf16 v16bf __attribute__((ext_vector_type(16)));
typedef __bf16 v4bf  __attribute__((ext_vector_type(4)));
typedef float  v8f   __attribute__((ext_vector_type(8)));

#define TAPS   512
#define MSUB   32
#define NB     16
#define TLEN   524288
#define SREAL  16385
#define SPAD   16640                 // 65 * 256 columns computed
#define PADR   16                    // zero rows in front of y for q-d < 0
#define ROWS   (PADR + SPAD)         // 16656
#define OUTB   (SREAL * MSUB)        // 524320 output samples per batch
#define BLOCKS_PER_B 65
#define BCOLS  256                   // columns per block (8 waves x 32)
#define XROWS  271                   // 256 block columns + 15 k-chunk overhang
#define FRAGN  (MSUB * TAPS)         // 16384 bf16 elements per fragment array

// ---- bf16 split: f = hi + lo (hi via native cvt, lo = remainder) ----
struct bfpair { __bf16 hi, lo; };
__device__ __forceinline__ bfpair splitbf(float f) {
    bfpair r;
    r.hi = (__bf16)f;
    r.lo = (__bf16)(f - (float)r.hi);
    return r;
}

__device__ __forceinline__ v16bf ldsv16(const __bf16* p) {
    v16bf r;
    __builtin_memcpy(&r, __builtin_assume_aligned(p, 32), 32);
    return r;
}

__device__ __forceinline__ v8f wmma_bf16(v16bf a, v16bf b, v8f c) {
    return __builtin_amdgcn_wmma_f32_16x16x32_bf16(false, a, false, b,
                                                   (short)0, c, false, false);
}

// Fragment-order permutation for the WMMA A operand (this code's contiguous k order, B matches):
// lane-half h holds K elements {h*16+0..15};
//   h = koff>>4 ; slot = koff&15
__device__ __forceinline__ void frag_slot(int koff, int& h, int& e) {
    h = (koff >> 4) & 1;
    e = koff & 15;
}

// ---------------- prep: split + permute filters, zero y pad rows ----------------
__global__ __launch_bounds__(256) void pqmf_prep(const float* __restrict__ h,
                                                 __bf16* __restrict__ afh,
                                                 __bf16* __restrict__ afl,
                                                 __bf16* __restrict__ sfh,
                                                 __bf16* __restrict__ sfl,
                                                 float* __restrict__ yt) {
    int i = blockIdx.x * 256 + threadIdx.x;
    if (i < MSUB * TAPS) {
        const int m = i / TAPS, t = i % TAPS;
        const float f = h[i];
        const bfpair ab = splitbf(f);
        // analysis A: row m, K = t = kc*32 + koff
        {
            const int kc = t >> 5, koff = t & 31;
            int hh, e; frag_slot(koff, hh, e);
            const int idx = ((m * 16 + kc) * 2 + hh) * 16 + e;
            afh[idx] = ab.hi; afl[idx] = ab.lo;
        }
        // synthesis A: row t (t = r + 32*d), K = m ; gain 32 folded in
        {
            const bfpair cd = splitbf(f * (float)MSUB);
            int hh, e; frag_slot(m, hh, e);
            const int idx = (t * 2 + hh) * 16 + e;
            sfh[idx] = cd.hi; sfl[idx] = cd.lo;
        }
    }
    if (i < NB * PADR * MSUB) {
        int bb = i / (PADR * MSUB);
        int r  = i % (PADR * MSUB);
        yt[(size_t)bb * ROWS * MSUB + r] = 0.0f;
    }
}

// ---------------- analysis: Y[b][PADR+s][m] = sum_t h[m,t] x[s*32+t] ----------------
__global__ __launch_bounds__(256) void pqmf_analysis(const float* __restrict__ x,
                                                     const __bf16* __restrict__ afh,
                                                     const __bf16* __restrict__ afl,
                                                     float* __restrict__ yt) {
    __shared__ __attribute__((aligned(32))) __bf16 sAh[FRAGN];
    __shared__ __attribute__((aligned(32))) __bf16 sAl[FRAGN];
    __shared__ __attribute__((aligned(32))) __bf16 sxh[XROWS * MSUB];
    __shared__ __attribute__((aligned(32))) __bf16 sxl[XROWS * MSUB];

    const int tid  = threadIdx.x;
    const int b    = blockIdx.x / BLOCKS_PER_B;
    const int tile = blockIdx.x % BLOCKS_PER_B;
    const int s0   = tile * BCOLS;

    { // stage fragment-ordered split filters (64 KB)
        const unsigned* srcH = (const unsigned*)afh;
        const unsigned* srcL = (const unsigned*)afl;
        unsigned* dH = (unsigned*)sAh;
        unsigned* dL = (unsigned*)sAl;
        for (int i = tid; i < FRAGN / 2; i += 256) { dH[i] = srcH[i]; dL[i] = srcL[i]; }
    }
    { // stage x slab pre-split to bf16 hi/lo, zero-extended past TLEN
        const float* xb  = x + (size_t)b * TLEN;
        const int   base = s0 * MSUB;
        for (int i4 = tid; i4 < (XROWS * MSUB) / 4; i4 += 256) {
            const int gi = base + i4 * 4;
            float4 v;
            if (gi + 3 < TLEN) v = *(const float4*)(xb + gi);
            else {
                v.x = (gi + 0 < TLEN) ? xb[gi + 0] : 0.f;
                v.y = (gi + 1 < TLEN) ? xb[gi + 1] : 0.f;
                v.z = (gi + 2 < TLEN) ? xb[gi + 2] : 0.f;
                v.w = (gi + 3 < TLEN) ? xb[gi + 3] : 0.f;
            }
            const bfpair p0 = splitbf(v.x), p1 = splitbf(v.y);
            const bfpair p2 = splitbf(v.z), p3 = splitbf(v.w);
            v4bf hv, lv;
            hv[0] = p0.hi; lv[0] = p0.lo;
            hv[1] = p1.hi; lv[1] = p1.lo;
            hv[2] = p2.hi; lv[2] = p2.lo;
            hv[3] = p3.hi; lv[3] = p3.lo;
            *(v4bf*)&sxh[i4 * 4] = hv;
            *(v4bf*)&sxl[i4 * 4] = lv;
        }
    }
    __syncthreads();

    const int lane  = tid & 31;
    const int w     = tid >> 5;
    const int n     = lane & 15;
    const int hi16  = lane >> 4;
    const int sub16 = hi16 * 16;
    const int colA  = w * 32 + n;     // first 16-column group of this wave
    const int r0    = lane & 15;      // A row within tile

    // hot-loop base pointers (advance by constant stride per kc)
    const __bf16* pBhA = &sxh[colA * MSUB + sub16];
    const __bf16* pBlA = &sxl[colA * MSUB + sub16];
    const __bf16* pBhB = pBhA + 16 * MSUB;   // second column group (+16 cols)
    const __bf16* pBlB = pBlA + 16 * MSUB;
    const int aBase0   = r0 * 512 + hi16 * 16;   // tile0 frag offset, +32/kc
    const int aBase1   = aBase0 + 16 * 512;      // tile1

    v8f acc0a = {}; v8f acc1a = {};   // m 0-15 / 16-31, columns colA..colA+15
    v8f acc0b = {}; v8f acc1b = {};   // columns colA+16..colA+31

#pragma unroll 4
    for (int kc = 0; kc < 16; ++kc) {
        const v16bf bhA = ldsv16(pBhA + kc * MSUB);
        const v16bf blA = ldsv16(pBlA + kc * MSUB);
        const v16bf bhB = ldsv16(pBhB + kc * MSUB);
        const v16bf blB = ldsv16(pBlB + kc * MSUB);

        const v16bf ah0 = ldsv16(&sAh[aBase0 + kc * 32]);
        const v16bf al0 = ldsv16(&sAl[aBase0 + kc * 32]);
        const v16bf ah1 = ldsv16(&sAh[aBase1 + kc * 32]);
        const v16bf al1 = ldsv16(&sAl[aBase1 + kc * 32]);

        acc0a = wmma_bf16(ah0, bhA, acc0a);
        acc0a = wmma_bf16(ah0, blA, acc0a);
        acc0a = wmma_bf16(al0, bhA, acc0a);
        acc1a = wmma_bf16(ah1, bhA, acc1a);
        acc1a = wmma_bf16(ah1, blA, acc1a);
        acc1a = wmma_bf16(al1, bhA, acc1a);

        acc0b = wmma_bf16(ah0, bhB, acc0b);
        acc0b = wmma_bf16(ah0, blB, acc0b);
        acc0b = wmma_bf16(al0, bhB, acc0b);
        acc1b = wmma_bf16(ah1, bhB, acc1b);
        acc1b = wmma_bf16(ah1, blB, acc1b);
        acc1b = wmma_bf16(al1, bhB, acc1b);
    }

    // D layout: lane holds column N = n, rows m = v + 8*hi16 (+16 for tile 1)
    float* ybA = yt + ((size_t)b * ROWS + (PADR + s0 + colA)) * MSUB;
    __builtin_memcpy(ybA + hi16 * 8, &acc0a, 32);
    __builtin_memcpy(ybA + 16 + hi16 * 8, &acc1a, 32);
    float* ybB = ybA + (size_t)16 * MSUB;
    __builtin_memcpy(ybB + hi16 * 8, &acc0b, 32);
    __builtin_memcpy(ybB + 16 + hi16 * 8, &acc1b, 32);
}

// ------- synthesis: out[b,q*32+r] = 32*sum_{m,d} h[m,r+32d]*y[b,m,q-d] -------
__global__ __launch_bounds__(256) void pqmf_synthesis(const float* __restrict__ yt,
                                                      const __bf16* __restrict__ sfh,
                                                      const __bf16* __restrict__ sfl,
                                                      float* __restrict__ out) {
    __shared__ __attribute__((aligned(32))) __bf16 sWh[FRAGN];
    __shared__ __attribute__((aligned(32))) __bf16 sWl[FRAGN];
    __shared__ __attribute__((aligned(32))) __bf16 syh[XROWS * MSUB];
    __shared__ __attribute__((aligned(32))) __bf16 syl[XROWS * MSUB];

    const int tid  = threadIdx.x;
    const int b    = blockIdx.x / BLOCKS_PER_B;
    const int tile = blockIdx.x % BLOCKS_PER_B;
    const int q0   = tile * BCOLS;

    { // stage fragment-ordered split filters (64 KB)
        const unsigned* srcH = (const unsigned*)sfh;
        const unsigned* srcL = (const unsigned*)sfl;
        unsigned* dH = (unsigned*)sWh;
        unsigned* dL = (unsigned*)sWl;
        for (int i = tid; i < FRAGN / 2; i += 256) { dH[i] = srcH[i]; dL[i] = srcL[i]; }
    }
    { // stage y rows [q0-15 .. q0+255], pre-split (pad rows keep this in-bounds)
        const int gbase = PADR + q0 - 15;          // >= 1
        const float4* src = (const float4*)(yt + ((size_t)b * ROWS + gbase) * MSUB);
        for (int i4 = tid; i4 < (XROWS * MSUB) / 4; i4 += 256) {
            const float4 v = src[i4];
            const bfpair p0 = splitbf(v.x), p1 = splitbf(v.y);
            const bfpair p2 = splitbf(v.z), p3 = splitbf(v.w);
            v4bf hv, lv;
            hv[0] = p0.hi; lv[0] = p0.lo;
            hv[1] = p1.hi; lv[1] = p1.lo;
            hv[2] = p2.hi; lv[2] = p2.lo;
            hv[3] = p3.hi; lv[3] = p3.lo;
            *(v4bf*)&syh[i4 * 4] = hv;
            *(v4bf*)&syl[i4 * 4] = lv;
        }
    }
    __syncthreads();

    const int lane  = tid & 31;
    const int w     = tid >> 5;
    const int n     = lane & 15;
    const int hi16  = lane >> 4;
    const int sub16 = hi16 * 16;
    const int r0    = lane & 15;

    // B rows run backwards in kc: rowA = (w*32 + n + 15) - kc ; rowB = rowA + 16
    const __bf16* pBhA = &syh[(w * 32 + n + 15) * MSUB + sub16];
    const __bf16* pBlA = &syl[(w * 32 + n + 15) * MSUB + sub16];
    const __bf16* pBhB = pBhA + 16 * MSUB;
    const __bf16* pBlB = pBlA + 16 * MSUB;
    const int aBase0   = r0 * 32 + hi16 * 16;    // t = r0 + 32*kc -> +1024/kc
    const int aBase1   = aBase0 + 16 * 32;       // t = r0 + 16 + 32*kc

    v8f acc0a = {}; v8f acc1a = {};
    v8f acc0b = {}; v8f acc1b = {};

#pragma unroll 4
    for (int kc = 0; kc < 16; ++kc) {
        const v16bf bhA = ldsv16(pBhA - kc * MSUB);
        const v16bf blA = ldsv16(pBlA - kc * MSUB);
        const v16bf bhB = ldsv16(pBhB - kc * MSUB);
        const v16bf blB = ldsv16(pBlB - kc * MSUB);

        const v16bf ah0 = ldsv16(&sWh[aBase0 + kc * 1024]);
        const v16bf al0 = ldsv16(&sWl[aBase0 + kc * 1024]);
        const v16bf ah1 = ldsv16(&sWh[aBase1 + kc * 1024]);
        const v16bf al1 = ldsv16(&sWl[aBase1 + kc * 1024]);

        acc0a = wmma_bf16(ah0, bhA, acc0a);
        acc0a = wmma_bf16(ah0, blA, acc0a);
        acc0a = wmma_bf16(al0, bhA, acc0a);
        acc1a = wmma_bf16(ah1, bhA, acc1a);
        acc1a = wmma_bf16(ah1, blA, acc1a);
        acc1a = wmma_bf16(al1, bhA, acc1a);

        acc0b = wmma_bf16(ah0, bhB, acc0b);
        acc0b = wmma_bf16(ah0, blB, acc0b);
        acc0b = wmma_bf16(al0, bhB, acc0b);
        acc1b = wmma_bf16(ah1, bhB, acc1b);
        acc1b = wmma_bf16(ah1, blB, acc1b);
        acc1b = wmma_bf16(al1, bhB, acc1b);
    }

    const int qA = q0 + w * 32 + n;
    if (qA < SREAL) {
        float* ob = out + (size_t)b * OUTB + (size_t)qA * MSUB;
        __builtin_memcpy(ob + hi16 * 8, &acc0a, 32);
        __builtin_memcpy(ob + 16 + hi16 * 8, &acc1a, 32);
    }
    const int qB = qA + 16;
    if (qB < SREAL) {
        float* ob = out + (size_t)b * OUTB + (size_t)qB * MSUB;
        __builtin_memcpy(ob + hi16 * 8, &acc0b, 32);
        __builtin_memcpy(ob + 16 + hi16 * 8, &acc1b, 32);
    }
}

extern "C" void kernel_launch(void* const* d_in, const int* in_sizes, int n_in,
                              void* d_out, int out_size, void* d_ws, size_t ws_size,
                              hipStream_t stream) {
    const float* x = (const float*)d_in[0];
    const float* h = (const float*)d_in[1];
    // d_in[2] (g) unused: g[m,k] == h[m,511-k]; synthesis derived from h.
    float* out = (float*)d_out;

    char*  ws = (char*)d_ws;
    float* yt = (float*)ws;                                   // [NB][ROWS][32] f32
    size_t ytBytes = (size_t)NB * ROWS * MSUB * sizeof(float);
    __bf16* afh = (__bf16*)(ws + ytBytes);                    // analysis frags hi
    __bf16* afl = afh + FRAGN;                                // analysis frags lo
    __bf16* sfh = afl + FRAGN;                                // synthesis frags hi
    __bf16* sfl = sfh + FRAGN;                                // synthesis frags lo

    pqmf_prep<<<(MSUB * TAPS + 255) / 256, 256, 0, stream>>>(h, afh, afl, sfh, sfl, yt);
    pqmf_analysis<<<NB * BLOCKS_PER_B, 256, 0, stream>>>(x, afh, afl, yt);
    pqmf_synthesis<<<NB * BLOCKS_PER_B, 256, 0, stream>>>(yt, sfh, sfl, out);
}
